// ChordDecoder_46832323396325
// MI455X (gfx1250) — compile-verified, matrix operand order checked by reading, not run on hardware
//
#include <hip/hip_runtime.h>

typedef __attribute__((ext_vector_type(16))) __bf16 v16bf;
typedef __attribute__((ext_vector_type(8)))  float  v8f;
typedef __attribute__((ext_vector_type(4)))  unsigned int u32x4;

#define BSZ   32768
#define HD    512
#define ZD    256
#define TOKD  36
#define NSTEP 8
#define XK    320      // packed x width: z_in(256) + token(36) + zero pad(28)
#define BM    64       // batch rows per workgroup (4 row tiles of 16)

// LDS row strides (elements); bf16 row pitch must be a multiple of 16 bytes
#define HS    528      // bf16 h rows
#define XS    336      // bf16 x rows
#define HEADS 68       // f32 head rows

// ---------------------------------------------------------------------------
// Fragment loader: 16-bit WMMA operand layout (both A 16x32 and B 32x16):
// lane L holds row/col (L & 15); K = {8*half .. 8*half+7} and {16+8*half ..}
// => two aligned 16B loads per lane.
// ---------------------------------------------------------------------------
__device__ __forceinline__ v16bf ld_frag(const __bf16* p, int hv) {
  union { u32x4 q[2]; v16bf v; } u;
  const __bf16* q = p + hv * 8;
  u.q[0] = *(const u32x4*)(q);
  u.q[1] = *(const u32x4*)(q + 16);
  return u.v;
}

__device__ __forceinline__ v8f wmma_bf16(v16bf a, v16bf b, v8f c) {
  return __builtin_amdgcn_wmma_f32_16x16x32_bf16(false, a, false, b, (short)0, c,
                                                 false, false);
}

__device__ __forceinline__ float sigmoidf_(float x) {
  return 1.0f / (1.0f + __expf(-x));
}

// ---------------------------------------------------------------------------
// Prep kernels: f32 -> bf16 weight repacks
// ---------------------------------------------------------------------------
__global__ void prep_cvt(const float* __restrict__ s, __bf16* __restrict__ d, int n) {
  int i = blockIdx.x * 256 + threadIdx.x;
  if (i < n) d[i] = (__bf16)s[i];
}

// W_ih is [1536][292] with columns [token(36) | z_in(256)].
// Repack to [1536][320] with columns [z_in(256) | token(36) | 0-pad(28)].
__global__ void prep_wih(const float* __restrict__ s, __bf16* __restrict__ d) {
  int i = blockIdx.x * 256 + threadIdx.x;
  if (i >= 3 * HD * XK) return;
  int j = i / XK, k = i % XK;
  float v = 0.0f;
  if (k < ZD)             v = s[j * 292 + TOKD + k];
  else if (k < ZD + TOKD) v = s[j * 292 + (k - ZD)];
  d[i] = (__bf16)v;
}

// Fuse root(12) / chroma(24) / bass(12) heads into one [64][512] bf16 matrix.
__global__ void prep_whead(const float* __restrict__ rw, const float* __restrict__ cw,
                           const float* __restrict__ bw, __bf16* __restrict__ d) {
  int i = blockIdx.x * 256 + threadIdx.x;
  if (i >= 64 * HD) return;
  int j = i / HD, k = i % HD;
  float v = 0.0f;
  if (j < 12)      v = rw[j * HD + k];
  else if (j < 36) v = cw[(j - 12) * HD + k];
  else if (j < 48) v = bw[(j - 36) * HD + k];
  d[i] = (__bf16)v;
}

__global__ void prep_bias(const float* __restrict__ b_ih, const float* __restrict__ b_hh,
                          const float* __restrict__ root_b, const float* __restrict__ chroma_b,
                          const float* __restrict__ bass_b,
                          float* __restrict__ bR, float* __restrict__ bZ,
                          float* __restrict__ bIN, float* __restrict__ bHN,
                          float* __restrict__ bhead) {
  int i = blockIdx.x * 256 + threadIdx.x;
  if (i < HD) {
    bR[i]  = b_ih[i] + b_hh[i];
    bZ[i]  = b_ih[HD + i] + b_hh[HD + i];
    bIN[i] = b_ih[2 * HD + i];
    bHN[i] = b_hh[2 * HD + i];
  }
  if (i < 64) {
    float v = 0.0f;
    if (i < 12)      v = root_b[i];
    else if (i < 36) v = chroma_b[i - 12];
    else if (i < 48) v = bass_b[i - 36];
    bhead[i] = v;
  }
}

// ---------------------------------------------------------------------------
// Persistent fused GRU decoder: one WG per 64 batch rows, 8 wave32.
// Each wave owns 16 h-columns per 128-column chunk; state lives in LDS.
// ---------------------------------------------------------------------------
__global__ __launch_bounds__(256)
void chord_decoder_main(
    const __bf16* __restrict__ zb,     // [BSZ][256] bf16 z_chd
    const __bf16* __restrict__ wih,    // [1536][320]
    const __bf16* __restrict__ whh,    // [1536][512]
    const __bf16* __restrict__ whid,   // [512][256]
    const __bf16* __restrict__ win,    // [256][256]
    const __bf16* __restrict__ whead,  // [64][512]
    const float* __restrict__ bRv, const float* __restrict__ bZv,
    const float* __restrict__ bINv, const float* __restrict__ bHNv,
    const float* __restrict__ bhid, const float* __restrict__ binb,
    const float* __restrict__ bheadv,
    const float* __restrict__ init_input,
    float* __restrict__ out_root,
    float* __restrict__ out_chroma,
    float* __restrict__ out_bass)
{
  extern __shared__ char smem[];
  __bf16* hB0   = (__bf16*)smem;             // [64][HS]  bf16 h, buffer 0
  __bf16* hB1   = hB0 + BM * HS;             // [64][HS]  bf16 h, buffer 1
  __bf16* xB    = hB1 + BM * HS;             // [64][XS]  bf16 x
  float*  headS = (float*)(xB + BM * XS);    // [64][HEADS] f32 head logits

  const int tid  = threadIdx.x;
  const int wv   = tid >> 5;
  const int lane = tid & 31;
  const int hv   = lane >> 4;   // which half of the wave
  const int idx  = lane & 15;   // row (A) / col (B/D) index within tile
  const int r0   = blockIdx.x * BM;

  // ---------------- initial GEMMs: h0 = z@Whid^T+b, z_in = z@Win^T+b --------
  #pragma unroll
  for (int c = 0; c < 4; ++c) {
    const int ct = c * 8 + wv;                       // column tile of H
    const float bias = bhid[ct * 16 + idx];
    #pragma unroll
    for (int rt = 0; rt < 4; ++rt) {
      v8f acc;
      #pragma unroll
      for (int i = 0; i < 8; ++i) acc[i] = bias;
      const __bf16* arow = zb + (size_t)(r0 + rt * 16 + idx) * ZD;
      const __bf16* brow = whid + (size_t)(ct * 16 + idx) * ZD;
      for (int k0 = 0; k0 < ZD; k0 += 32)
        acc = wmma_bf16(ld_frag(arow + k0, hv), ld_frag(brow + k0, hv), acc);
      #pragma unroll
      for (int r = 0; r < 8; ++r) {
        const int M = rt * 16 + r + 8 * hv;
        hB0[M * HS + ct * 16 + idx] = (__bf16)acc[r];
      }
    }
  }
  #pragma unroll
  for (int c = 0; c < 2; ++c) {
    const int ct = c * 8 + wv;                       // column tile of ZIN
    const float bias = binb[ct * 16 + idx];
    #pragma unroll
    for (int rt = 0; rt < 4; ++rt) {
      v8f acc;
      #pragma unroll
      for (int i = 0; i < 8; ++i) acc[i] = bias;
      const __bf16* arow = zb + (size_t)(r0 + rt * 16 + idx) * ZD;
      const __bf16* brow = win + (size_t)(ct * 16 + idx) * ZD;
      for (int k0 = 0; k0 < ZD; k0 += 32)
        acc = wmma_bf16(ld_frag(arow + k0, hv), ld_frag(brow + k0, hv), acc);
      #pragma unroll
      for (int r = 0; r < 8; ++r) {
        const int M = rt * 16 + r + 8 * hv;
        xB[M * XS + ct * 16 + idx] = (__bf16)acc[r];
      }
    }
  }
  // token0 broadcast + zero K-padding
  if (tid < BM) {
    const int m = tid;
    for (int k = 0; k < TOKD; ++k) xB[m * XS + ZD + k] = (__bf16)init_input[k];
    for (int k = ZD + TOKD; k < XS; ++k) xB[m * XS + k] = (__bf16)0.0f;
  }
  __syncthreads();

  // ---------------- recurrent steps ----------------------------------------
  for (int t = 0; t < NSTEP; ++t) {
    const __bf16* hRead  = (t & 1) ? hB1 : hB0;
    __bf16*       hWrite = (t & 1) ? hB0 : hB1;

    for (int c = 0; c < 4; ++c) {               // 128-column chunks of H
      const int j0  = c * 128 + wv * 16;        // this wave's 16 columns
      const int col = j0 + idx;

      const __bf16* wr = wih + (size_t)col * XK;
      const __bf16* wz = wih + (size_t)(HD + col) * XK;
      const __bf16* wn = wih + (size_t)(2 * HD + col) * XK;
      const __bf16* ur = whh + (size_t)col * HD;
      const __bf16* uz = whh + (size_t)(HD + col) * HD;
      const __bf16* un = whh + (size_t)(2 * HD + col) * HD;
      // warm L2/L0 for the recurrent-weight rows used by the second K loop
      __builtin_prefetch(ur, 0, 1);
      __builtin_prefetch(uz, 0, 1);
      __builtin_prefetch(un, 0, 1);

      v8f aR[4], aZ[4], aIN[4], aHN[4];
      {
        const float vR = bRv[col], vZ = bZv[col], vIN = bINv[col], vHN = bHNv[col];
        #pragma unroll
        for (int rt = 0; rt < 4; ++rt) {
          #pragma unroll
          for (int i = 0; i < 8; ++i) {
            aR[rt][i] = vR; aZ[rt][i] = vZ; aIN[rt][i] = vIN; aHN[rt][i] = vHN;
          }
        }
      }
      // x contribution: i_r, i_z, i_n  (K = 320)
      #pragma unroll 2
      for (int k0 = 0; k0 < XK; k0 += 32) {
        v16bf br = ld_frag(wr + k0, hv);
        v16bf bz = ld_frag(wz + k0, hv);
        v16bf bn = ld_frag(wn + k0, hv);
        #pragma unroll
        for (int rt = 0; rt < 4; ++rt) {
          v16bf a = ld_frag(xB + (size_t)(rt * 16 + idx) * XS + k0, hv);
          aR[rt]  = wmma_bf16(a, br, aR[rt]);
          aZ[rt]  = wmma_bf16(a, bz, aZ[rt]);
          aIN[rt] = wmma_bf16(a, bn, aIN[rt]);
        }
      }
      // h contribution: h_r, h_z, h_n  (K = 512)
      #pragma unroll 2
      for (int k0 = 0; k0 < HD; k0 += 32) {
        v16bf br = ld_frag(ur + k0, hv);
        v16bf bz = ld_frag(uz + k0, hv);
        v16bf bn = ld_frag(un + k0, hv);
        #pragma unroll
        for (int rt = 0; rt < 4; ++rt) {
          v16bf a = ld_frag(hRead + (size_t)(rt * 16 + idx) * HS + k0, hv);
          aR[rt]  = wmma_bf16(a, br, aR[rt]);
          aZ[rt]  = wmma_bf16(a, bz, aZ[rt]);
          aHN[rt] = wmma_bf16(a, bn, aHN[rt]);
        }
      }
      // gates + h update; (M,col) ownership is exclusive to this lane/element
      #pragma unroll
      for (int rt = 0; rt < 4; ++rt) {
        #pragma unroll
        for (int r = 0; r < 8; ++r) {
          const int M = rt * 16 + r + 8 * hv;
          const float rs = sigmoidf_(aR[rt][r]);
          const float zs = sigmoidf_(aZ[rt][r]);
          const float nn = tanhf(aIN[rt][r] + rs * aHN[rt][r]);
          const float hold = (float)hRead[M * HS + col];
          const float hnew = (1.0f - zs) * nn + zs * hold;
          hWrite[M * HS + col] = (__bf16)hnew;
        }
      }
    }
    __syncthreads();

    // head GEMM on new h: 4x4 tiles, two per wave
    #pragma unroll
    for (int rt2 = 0; rt2 < 2; ++rt2) {
      const int rt = (wv >> 2) + 2 * rt2;
      const int ct = wv & 3;
      const float bias = bheadv[ct * 16 + idx];
      v8f acc;
      #pragma unroll
      for (int i = 0; i < 8; ++i) acc[i] = bias;
      const __bf16* arow = hWrite + (size_t)(rt * 16 + idx) * HS;
      const __bf16* brow = whead + (size_t)(ct * 16 + idx) * HD;
      for (int k0 = 0; k0 < HD; k0 += 32)
        acc = wmma_bf16(ld_frag(arow + k0, hv), ld_frag(brow + k0, hv), acc);
      #pragma unroll
      for (int r = 0; r < 8; ++r) {
        const int M = rt * 16 + r + 8 * hv;
        headS[M * HEADS + ct * 16 + idx] = acc[r];
      }
    }
    __syncthreads();

    // write logits + argmax -> next one-hot token (one thread per batch row)
    if (tid < BM) {
      const int m = tid;
      const size_t b = (size_t)r0 + m;
      float v[48];
      #pragma unroll
      for (int i = 0; i < 48; ++i) v[i] = headS[m * HEADS + i];
      float* orp = out_root   + (b * NSTEP + t) * 12;
      float* ocp = out_chroma + (b * NSTEP + t) * 24;
      float* obp = out_bass   + (b * NSTEP + t) * 12;
      #pragma unroll
      for (int i = 0; i < 12; ++i) orp[i] = v[i];
      #pragma unroll
      for (int i = 0; i < 24; ++i) ocp[i] = v[12 + i];
      #pragma unroll
      for (int i = 0; i < 12; ++i) obp[i] = v[36 + i];
      int ar = 0; float mr = v[0];
      #pragma unroll
      for (int i = 1; i < 12; ++i) if (v[i] > mr) { mr = v[i]; ar = i; }
      int ab = 0; float mb = v[36];
      #pragma unroll
      for (int i = 1; i < 12; ++i) if (v[36 + i] > mb) { mb = v[36 + i]; ab = i; }
      #pragma unroll
      for (int i = 0; i < 12; ++i)
        xB[m * XS + ZD + i] = (__bf16)(i == ar ? 1.0f : 0.0f);
      #pragma unroll
      for (int p = 0; p < 12; ++p)
        xB[m * XS + ZD + 12 + p] = (__bf16)((v[12 + 2 * p + 1] > v[12 + 2 * p]) ? 1.0f : 0.0f);
      #pragma unroll
      for (int i = 0; i < 12; ++i)
        xB[m * XS + ZD + 24 + i] = (__bf16)(i == ab ? 1.0f : 0.0f);
    }
    __syncthreads();
  }
}

// ---------------------------------------------------------------------------
extern "C" void kernel_launch(void* const* d_in, const int* in_sizes, int n_in,
                              void* d_out, int out_size, void* d_ws, size_t ws_size,
                              hipStream_t stream) {
  (void)in_sizes; (void)n_in; (void)out_size; (void)ws_size;
  const float* z_chd    = (const float*)d_in[0];
  const float* init_inp = (const float*)d_in[1];
  const float* hidw     = (const float*)d_in[2];
  const float* hidb     = (const float*)d_in[3];
  const float* inw      = (const float*)d_in[4];
  const float* inb      = (const float*)d_in[5];
  const float* W_ih     = (const float*)d_in[6];
  const float* W_hh     = (const float*)d_in[7];
  const float* b_ih     = (const float*)d_in[8];
  const float* b_hh     = (const float*)d_in[9];
  const float* root_w   = (const float*)d_in[10];
  const float* root_b   = (const float*)d_in[11];
  const float* chroma_w = (const float*)d_in[12];
  const float* chroma_b = (const float*)d_in[13];
  const float* bass_w   = (const float*)d_in[14];
  const float* bass_b   = (const float*)d_in[15];

  char* ws = (char*)d_ws;
  size_t o = 0;
  auto take = [&](size_t bytes) -> void* {
    void* p = ws + o;
    o = (o + bytes + 255) & ~(size_t)255;
    return p;
  };
  __bf16* wih   = (__bf16*)take((size_t)3 * HD * XK * 2);
  __bf16* whh   = (__bf16*)take((size_t)3 * HD * HD * 2);
  __bf16* whid  = (__bf16*)take((size_t)HD * ZD * 2);
  __bf16* win   = (__bf16*)take((size_t)ZD * ZD * 2);
  __bf16* whead = (__bf16*)take((size_t)64 * HD * 2);
  __bf16* zb    = (__bf16*)take((size_t)BSZ * ZD * 2);
  float* bR    = (float*)take(HD * 4);
  float* bZ    = (float*)take(HD * 4);
  float* bIN   = (float*)take(HD * 4);
  float* bHN   = (float*)take(HD * 4);
  float* bhead = (float*)take(64 * 4);

  prep_wih<<<(3 * HD * XK + 255) / 256, 256, 0, stream>>>(W_ih, wih);
  prep_cvt<<<(3 * HD * HD + 255) / 256, 256, 0, stream>>>(W_hh, whh, 3 * HD * HD);
  prep_cvt<<<(HD * ZD + 255) / 256, 256, 0, stream>>>(hidw, whid, HD * ZD);
  prep_cvt<<<(ZD * ZD + 255) / 256, 256, 0, stream>>>(inw, win, ZD * ZD);
  prep_whead<<<(64 * HD + 255) / 256, 256, 0, stream>>>(root_w, chroma_w, bass_w, whead);
  prep_cvt<<<(BSZ * ZD + 255) / 256, 256, 0, stream>>>(z_chd, zb, BSZ * ZD);
  prep_bias<<<(HD + 255) / 256, 256, 0, stream>>>(b_ih, b_hh, root_b, chroma_b, bass_b,
                                                  bR, bZ, bIN, bHN, bhead);

  float* out_root   = (float*)d_out;
  float* out_chroma = out_root + (size_t)BSZ * NSTEP * 12;
  float* out_bass   = out_chroma + (size_t)BSZ * NSTEP * 24;

  const size_t lds = (size_t)BM * HS * 2 * 2   // hB0 + hB1 (bf16)
                   + (size_t)BM * XS * 2       // xB (bf16)
                   + (size_t)BM * HEADS * 4;   // headS (f32) => 195584 B
  chord_decoder_main<<<BSZ / BM, 256, lds, stream>>>(
      zb, wih, whh, whid, win, whead,
      bR, bZ, bIN, bHN, hidb, inb, bhead, init_inp,
      out_root, out_chroma, out_bass);
}
